// Net_15410342658439
// MI455X (gfx1250) — matrix operand order repaired and checked by reading, compiled, not run on hardware
//
#include <hip/hip_runtime.h>
#include <hip/hip_bf16.h>

// ---------------------------------------------------------------------------
// GCN 2-layer forward for MI455X (gfx1250, wave32).
//  GEMMs:   bf16 WMMA (v_wmma_f32_16x16x32_bf16), fp32 accumulate.
//  Scatter: float4 gathers + global_atomic_add_f32 (L2-resident working set).
// ---------------------------------------------------------------------------

typedef __attribute__((ext_vector_type(16))) __bf16 bf16x16;
typedef __attribute__((ext_vector_type(8)))  float  f32x8;

__device__ __forceinline__ unsigned short f32_to_bf16(float f) {
    union { float f; unsigned u; } x; x.f = f;
    unsigned r = x.u + 0x7FFFu + ((x.u >> 16) & 1u);   // round-to-nearest-even
    return (unsigned short)(r >> 16);
}
__device__ __forceinline__ unsigned pack_bf16x2(float a, float b) {
    return (unsigned)f32_to_bf16(a) | ((unsigned)f32_to_bf16(b) << 16);
}

// ---------------------------------------------------------------------------
// Tiled WMMA GEMM:  C[M,Ntot] = A[M,K] @ B[K,Ntot]   (A,B fp32 -> bf16)
// Workgroup: 256 threads = 8 waves. Each wave owns a 16-row M tile
// (wave w -> rows [blockIdx.x*128 + 16w, +16)) and NACC 16-wide N tiles.
// blockIdx.y selects a chunk of NACC*16 columns.
// ---------------------------------------------------------------------------
template <int NACC>
__global__ __launch_bounds__(256)
void gemm_bf16_wmma(const float* __restrict__ A, const float* __restrict__ B,
                    float* __restrict__ C, int M, int K, int Ntot, int ldc) {
    constexpr int NCHUNK = NACC * 16;
    __shared__ unsigned Ald[128][17];     // 128 rows x 32 bf16 (16 dwords) +pad
    __shared__ unsigned Bld[NCHUNK][17];  // NCHUNK cols x 32 bf16 (16 dwords) +pad

    const int m0   = blockIdx.x * 128;
    const int nb   = blockIdx.y * NCHUNK;
    const int tid  = threadIdx.x;
    const int wid  = tid >> 5;
    const int lane = tid & 31;

    f32x8 acc[NACC] = {};

    for (int k0 = 0; k0 < K; k0 += 32) {
        // ---- stage A tile: 128 rows x 16 dwords (2 bf16 each) ----
        for (int i = tid; i < 128 * 16; i += 256) {
            const int row = i >> 4, j = i & 15;
            const int m = m0 + row;
            float ax = 0.f, ay = 0.f;
            if (m < M) {
                const float2 t = *(const float2*)(A + (long long)m * K + k0 + 2 * j);
                ax = t.x; ay = t.y;
            }
            Ald[row][j] = pack_bf16x2(ax, ay);
        }
        // ---- stage B tile (column-major in LDS: Bld[n][k_pair]) ----
        for (int i = tid; i < NCHUNK * 16; i += 256) {
            const int j = i / NCHUNK;       // k-pair index 0..15
            const int n = i % NCHUNK;
            const int col = nb + n;
            float b0 = 0.f, b1 = 0.f;
            if (col < Ntot) {
                b0 = B[(long long)(k0 + 2 * j) * Ntot + col];
                b1 = B[(long long)(k0 + 2 * j + 1) * Ntot + col];
            }
            Bld[n][j] = pack_bf16x2(b0, b1);
        }
        __syncthreads();

        // ---- A fragment (ISA 7.12.2 16-bit A 16x32 layout) ----
        // lane&15 = M row; lanes<16 hold K 0-7,16-23; lanes>=16 hold K 8-15,24-31
        const int arow = wid * 16 + (lane & 15);
        const int aoff = (lane >> 4) * 4;
        union { bf16x16 v; unsigned u[8]; } af;
#pragma unroll
        for (int j = 0; j < 4; ++j) {
            af.u[j]     = Ald[arow][aoff + j];
            af.u[4 + j] = Ald[arow][aoff + 8 + j];
        }
        // ---- B fragments: lane&15 = N col; lanes<16 K=0-7,16-23; lanes>=16 K=8-15,24-31
        const int boff = (lane >> 4) * 4;
#pragma unroll
        for (int a = 0; a < NACC; ++a) {
            union { bf16x16 v; unsigned u[8]; } bf;
            const int n = a * 16 + (lane & 15);
#pragma unroll
            for (int j = 0; j < 8; ++j) bf.u[j] = Bld[n][boff + (j & 3) + (j >> 2) * 8];
            acc[a] = __builtin_amdgcn_wmma_f32_16x16x32_bf16(
                false, af.v, false, bf.v, (short)0, acc[a], false, false);
        }
        __syncthreads();
    }

    // ---- store: C/D layout — lanes<16: N=lane, M=r; lanes>=16: N=lane-16, M=r+8
    const int mbase = m0 + wid * 16 + ((lane >> 4) << 3);
    const int ncol0 = nb + (lane & 15);
#pragma unroll
    for (int a = 0; a < NACC; ++a) {
#pragma unroll
        for (int r = 0; r < 8; ++r) {
            const int m = mbase + r;
            const int n = ncol0 + a * 16;
            if (m < M && n < Ntot) C[(long long)m * ldc + n] = acc[a][r];
        }
    }
}

// ---------------------------------------------------------------------------
// Degree / normalization helpers
// ---------------------------------------------------------------------------
__global__ void k_zero_f32(float* __restrict__ p, long long n) {
    long long i = (long long)blockIdx.x * blockDim.x + threadIdx.x;
    if (i < n) p[i] = 0.f;
}
__global__ void k_deg_count(const int* __restrict__ dst, float* __restrict__ deg, int E) {
    int i = blockIdx.x * blockDim.x + threadIdx.x;
    if (i < E) atomicAdd(&deg[dst[i]], 1.0f);
}
__global__ void k_deg_to_dinv(float* __restrict__ deg, int n) {
    int i = blockIdx.x * blockDim.x + threadIdx.x;
    if (i < n) deg[i] = rsqrtf(deg[i] + 1.0f);   // +1 self-loop; always > 0
}

// out[i,f] = H[i,f] * dinv[i]^2   (self-loop contribution, initializes out)
__global__ void k_self_init(const float* __restrict__ H, const float* __restrict__ dinv,
                            float* __restrict__ out, long long total, int F) {
    long long i = (long long)blockIdx.x * blockDim.x + threadIdx.x;
    if (i < total) {
        const int node = (int)(i / F);
        const float d = dinv[node];
        out[i] = H[i] * d * d;
    }
}

// Edge scatter: out[dst,f] += H[src,f] * dinv[src]*dinv[dst], float4 per thread
template <int F>
__global__ void k_scatter_edges(const float* __restrict__ H, const float* __restrict__ dinv,
                                const int* __restrict__ src, const int* __restrict__ dst,
                                float* __restrict__ out, long long totalQuads) {
    constexpr int Q = F / 4;
    long long idx = (long long)blockIdx.x * blockDim.x + threadIdx.x;
    if (idx >= totalQuads) return;
    const int e = (int)(idx / Q);
    const int q = (int)(idx % Q);
    const int s = src[e];
    const int d = dst[e];
    const float nrm = dinv[s] * dinv[d];
    const float4 v = *(const float4*)(H + (long long)s * F + 4 * q);
    float* o = out + (long long)d * F + 4 * q;
    atomicAdd(o + 0, v.x * nrm);
    atomicAdd(o + 1, v.y * nrm);
    atomicAdd(o + 2, v.z * nrm);
    atomicAdd(o + 3, v.w * nrm);
}

__global__ void k_bias_relu(float* __restrict__ x, const float* __restrict__ b,
                            long long total, int F) {
    long long i = (long long)blockIdx.x * blockDim.x + threadIdx.x;
    if (i < total) x[i] = fmaxf(x[i] + b[(int)(i % F)], 0.f);
}

// Per-row (40-wide) bias + log_softmax; one wave32 per row, shuffle reductions.
__global__ void k_bias_logsoftmax(float* __restrict__ out, const float* __restrict__ bias,
                                  int Nrows, int C) {
    const int row  = blockIdx.x * (blockDim.x >> 5) + (threadIdx.x >> 5);
    const int lane = threadIdx.x & 31;
    if (row >= Nrows) return;
    float* p = out + (long long)row * C;

    const bool v0 = (lane < C);
    const bool v1 = (lane + 32 < C);
    float a = v0 ? (p[lane] + bias[lane]) : -3.0e38f;
    float b = v1 ? (p[lane + 32] + bias[lane + 32]) : -3.0e38f;

    float m = fmaxf(a, b);
#pragma unroll
    for (int off = 16; off > 0; off >>= 1) m = fmaxf(m, __shfl_xor(m, off, 32));

    float e = (v0 ? __expf(a - m) : 0.f) + (v1 ? __expf(b - m) : 0.f);
#pragma unroll
    for (int off = 16; off > 0; off >>= 1) e += __shfl_xor(e, off, 32);

    const float lse = __logf(e) + m;
    if (v0) p[lane] = a - lse;
    if (v1) p[lane + 32] = b - lse;
}

// ---------------------------------------------------------------------------
// Launch
// ---------------------------------------------------------------------------
extern "C" void kernel_launch(void* const* d_in, const int* in_sizes, int n_in,
                              void* d_out, int out_size, void* d_ws, size_t ws_size,
                              hipStream_t stream) {
    (void)n_in; (void)out_size; (void)ws_size;

    const float* x  = (const float*)d_in[0];
    const float* W1 = (const float*)d_in[1];
    const float* b1 = (const float*)d_in[2];
    const float* W2 = (const float*)d_in[3];
    const float* b2 = (const float*)d_in[4];
    const int*   ei = (const int*)d_in[5];      // [2, E] int32 (jax default x64 off)

    const int Hdim = in_sizes[2];               // 256
    const int Cdim = in_sizes[4];               // 40
    const int Fin  = in_sizes[1] / Hdim;        // 512
    const int Nn   = in_sizes[0] / Fin;         // 100000
    const int E    = in_sizes[5] / 2;           // 3200000
    const int* src = ei;
    const int* dst = ei + E;

    // Workspace layout (floats):  dinv | H1 | O1 ; H2 overlays H1 (dead after ReLU)
    float* ws   = (float*)d_ws;
    float* dinv = ws;
    float* H1   = ws + (((long long)Nn + 511) / 512) * 512;
    float* O1   = H1 + (long long)Nn * Hdim;
    float* H2   = H1;                           // reuse: H1 dead once O1 is ReLU'd
    float* out  = (float*)d_out;

    const int T = 256;

    // 1) degrees -> dinv  (in-degree incl. self-loop)
    k_zero_f32<<<(Nn + T - 1) / T, T, 0, stream>>>(dinv, Nn);
    k_deg_count<<<(E + T - 1) / T, T, 0, stream>>>(dst, dinv, E);
    k_deg_to_dinv<<<(Nn + T - 1) / T, T, 0, stream>>>(dinv, Nn);

    // 2) H1 = X @ W1   (bf16 WMMA)
    {
        dim3 g((Nn + 127) / 128, Hdim / 64);
        gemm_bf16_wmma<4><<<g, T, 0, stream>>>(x, W1, H1, Nn, Fin, Hdim, Hdim);
    }

    // 3) O1 = Dn (A+I) Dn H1 ; then ReLU(O1 + b1)
    const long long tot1 = (long long)Nn * Hdim;
    k_self_init<<<(unsigned)((tot1 + T - 1) / T), T, 0, stream>>>(H1, dinv, O1, tot1, Hdim);
    {
        const long long q1 = (long long)E * (Hdim / 4);
        k_scatter_edges<256><<<(unsigned)((q1 + T - 1) / T), T, 0, stream>>>(
            H1, dinv, src, dst, O1, q1);
    }
    k_bias_relu<<<(unsigned)((tot1 + T - 1) / T), T, 0, stream>>>(O1, b1, tot1, Hdim);

    // 4) H2 = O1 @ W2   (bf16 WMMA, N=40 padded to 48 in-tile)
    {
        dim3 g((Nn + 127) / 128, 1);
        gemm_bf16_wmma<3><<<g, T, 0, stream>>>(O1, W2, H2, Nn, Hdim, Cdim, Cdim);
    }

    // 5) aggregate layer 2 directly into d_out
    const long long tot2 = (long long)Nn * Cdim;
    k_self_init<<<(unsigned)((tot2 + T - 1) / T), T, 0, stream>>>(H2, dinv, out, tot2, Cdim);
    {
        const long long q2 = (long long)E * (Cdim / 4);
        k_scatter_edges<40><<<(unsigned)((q2 + T - 1) / T), T, 0, stream>>>(
            H2, dinv, src, dst, out, q2);
    }

    // 6) out = log_softmax(out + b2), one wave per row
    k_bias_logsoftmax<<<(Nn + 7) / 8, T, 0, stream>>>(out, b2, Nn, Cdim);
}
